// ResidualAttentionBlock_13280038880040
// MI455X (gfx1250) — compile-verified, matrix operand order checked by reading, not run on hardware
//
#include <hip/hip_runtime.h>
#include <hip/hip_bf16.h>
#include <math.h>

// ---------------------------------------------------------------------------
// ResidualAttentionBlock (linear attention) for gfx1250 / MI455X.
// B=8, N=4096, C=512, H=16, D=32, HIDDEN=512. All GEMMs via
// v_wmma_f32_16x16x32_bf16 (bf16 inputs, f32 accumulation).
// GEMM tile staging uses gfx1250 async global->LDS copies (ASYNCcnt) with
// double buffering.
// ---------------------------------------------------------------------------

typedef __bf16 bf16_t;
typedef __attribute__((ext_vector_type(8)))  __bf16 bf16x8;
typedef __attribute__((ext_vector_type(16))) __bf16 bf16x16;
typedef __attribute__((ext_vector_type(8)))  float  f32x8;

#define BB 8
#define NN 4096
#define CC 512
#define HH 16
#define DD 32

__device__ inline f32x8 wmma_bf16(bf16x16 a, bf16x16 b, f32x8 c) {
  // D = A*B + C ; 16x16x32, f32 accum
  return __builtin_amdgcn_wmma_f32_16x16x32_bf16(
      /*neg_a=*/false, a, /*neg_b=*/false, b,
      /*c_mod=*/(short)0, c, /*reuse_a=*/false, /*reuse_b=*/false);
}

// Load one 16x32 bf16 A/B fragment from a row-major matrix with leading dim
// `ld` (elements). Per ISA layout: lanes 0-15 hold row (rowbase+lane), K
// chunks {0..7,16..23}; lanes 16-31 hold same rows, K chunks {8..15,24..31}.
__device__ inline bf16x16 load_frag(const bf16_t* __restrict__ p0, int ld,
                                    int rowbase, int k0, int lane) {
  const bf16_t* p = p0 + (size_t)(rowbase + (lane & 15)) * ld + k0 + ((lane >> 4) << 3);
  bf16x8 lo = *(const bf16x8*)(p);
  bf16x8 hi = *(const bf16x8*)(p + 16);
  return __builtin_shufflevector(lo, hi, 0,1,2,3,4,5,6,7,8,9,10,11,12,13,14,15);
}

// Async 16-byte global -> LDS copy (gfx1250 GLOBAL_LOAD_ASYNC_TO_LDS_B128).
// VDST VGPR holds the wave-relative LDS byte offset = low 32 bits of the
// generic pointer to the __shared__ object.
__device__ inline void async_copy_b128(const bf16_t* gaddr, bf16_t* laddr) {
  asm volatile("global_load_async_to_lds_b128 %0, %1, off"
               :
               : "v"((uint32_t)(uintptr_t)laddr),
                 "v"((uint64_t)(uintptr_t)gaddr)
               : "memory");
}
__device__ inline void wait_async_le4() {
  asm volatile("s_wait_asynccnt 0x4" ::: "memory");
}
__device__ inline void wait_async_0() {
  asm volatile("s_wait_asynccnt 0x0" ::: "memory");
}

// ---------------------------------------------------------------------------
// Weight prep: f32 (in,out) -> bf16 transposed (out,in). WTqkv = [Wq;Wk;Wv].
// ---------------------------------------------------------------------------
__global__ __launch_bounds__(256) void prep_weights_kernel(
    const float* __restrict__ Wq, const float* __restrict__ Wk,
    const float* __restrict__ Wv, const float* __restrict__ Wm,
    const float* __restrict__ W1, const float* __restrict__ W2,
    bf16_t* __restrict__ WTqkv, bf16_t* __restrict__ WmT,
    bf16_t* __restrict__ W1T, bf16_t* __restrict__ W2T) {
  int idx = blockIdx.x * 256 + threadIdx.x;   // 0 .. 512*512-1
  int o = idx >> 9, i = idx & 511;
  size_t src = (size_t)i * CC + o;
  size_t dst = (size_t)o * CC + i;
  WTqkv[dst]                 = (bf16_t)Wq[src];
  WTqkv[dst + 512 * 512]     = (bf16_t)Wk[src];
  WTqkv[dst + 2 * 512 * 512] = (bf16_t)Wv[src];
  WmT[dst] = (bf16_t)Wm[src];
  W1T[dst] = (bf16_t)W1[src];
  W2T[dst] = (bf16_t)W2[src];
}

// ---------------------------------------------------------------------------
// LayerNorm over rows of 512 f32 -> bf16. One wave32 per row, 8 rows/block.
// ---------------------------------------------------------------------------
__global__ __launch_bounds__(256) void ln_bf16_kernel(
    const float* __restrict__ x, const float* __restrict__ gamma,
    const float* __restrict__ beta, bf16_t* __restrict__ out) {
  int row  = blockIdx.x * 8 + (threadIdx.x >> 5);
  int lane = threadIdx.x & 31;
  const float* xr = x + (size_t)row * CC;
  float v[16];
  float s = 0.f;
#pragma unroll
  for (int i = 0; i < 4; ++i) {
    float4 t = *(const float4*)(xr + i * 128 + lane * 4);
    v[4 * i + 0] = t.x; v[4 * i + 1] = t.y; v[4 * i + 2] = t.z; v[4 * i + 3] = t.w;
    s += t.x + t.y + t.z + t.w;
  }
#pragma unroll
  for (int off = 16; off > 0; off >>= 1) s += __shfl_xor(s, off, 32);
  float mu = s * (1.f / 512.f);
  float s2 = 0.f;
#pragma unroll
  for (int i = 0; i < 16; ++i) { float d = v[i] - mu; s2 += d * d; }
#pragma unroll
  for (int off = 16; off > 0; off >>= 1) s2 += __shfl_xor(s2, off, 32);
  float rstd = rsqrtf(s2 * (1.f / 512.f) + 1e-6f);
  bf16_t* orow = out + (size_t)row * CC;
#pragma unroll
  for (int i = 0; i < 4; ++i) {
#pragma unroll
    for (int j = 0; j < 4; ++j) {
      int idx = i * 128 + lane * 4 + j;
      orow[idx] = (bf16_t)((v[4 * i + j] - mu) * rstd * gamma[idx] + beta[idx]);
    }
  }
}

// ---------------------------------------------------------------------------
// Generic tiled GEMM: D(row,col) = sum_k A[row,k] * BT[col,k], K=512.
// Block = 256 threads (8 waves), 128x128 tile, K-step 32, double-buffered
// async global->LDS staging. Wave w: rows [(w&1)*64,+64), cols [(w>>1)*32,+32).
// Epilogue functor invoked per output element.
// ---------------------------------------------------------------------------
template <typename Epi>
__global__ __launch_bounds__(256) void gemm_bf16_kernel(
    const bf16_t* __restrict__ A, const bf16_t* __restrict__ BT, int K, Epi epi) {
  __shared__ __align__(16) bf16_t sA[2][128 * 32];
  __shared__ __align__(16) bf16_t sB[2][128 * 32];
  int tid  = threadIdx.x;
  int wave = tid >> 5, lane = tid & 31;
  int m0 = blockIdx.x * 128;
  int n0 = blockIdx.y * 128;
  int wm = (wave & 1) * 64;
  int wn = (wave >> 1) * 32;

  // This thread's two 16B staging chunks per tile (512 chunks / 256 threads).
  int c0 = tid,       r0c = c0 >> 2, s0c = c0 & 3;
  int c1 = tid + 256, r1c = c1 >> 2, s1c = c1 & 3;

  auto issue = [&](int k0, int buf) {
    async_copy_b128(A  + (size_t)(m0 + r0c) * K + k0 + s0c * 8, &sA[buf][r0c * 32 + s0c * 8]);
    async_copy_b128(BT + (size_t)(n0 + r0c) * K + k0 + s0c * 8, &sB[buf][r0c * 32 + s0c * 8]);
    async_copy_b128(A  + (size_t)(m0 + r1c) * K + k0 + s1c * 8, &sA[buf][r1c * 32 + s1c * 8]);
    async_copy_b128(BT + (size_t)(n0 + r1c) * K + k0 + s1c * 8, &sB[buf][r1c * 32 + s1c * 8]);
  };

  f32x8 acc[4][2] = {};
  int nk = K >> 5;          // 16 K-steps
  issue(0, 0);
  for (int it = 0; it < nk; ++it) {
    if (it + 1 < nk) {
      issue((it + 1) << 5, (it + 1) & 1);  // prefetch next tile (other buffer)
      wait_async_le4();                    // wait for tile `it` (in-order)
    } else {
      wait_async_0();
    }
    __syncthreads();                       // publish tile `it` to all waves

    const bf16_t* cA = sA[it & 1];
    const bf16_t* cB = sB[it & 1];
    bf16x16 bf0 = load_frag(cB, 32, wn, 0, lane);
    bf16x16 bf1 = load_frag(cB, 32, wn + 16, 0, lane);
#pragma unroll
    for (int mi = 0; mi < 4; ++mi) {
      bf16x16 af = load_frag(cA, 32, wm + mi * 16, 0, lane);
      acc[mi][0] = wmma_bf16(af, bf0, acc[mi][0]);
      acc[mi][1] = wmma_bf16(af, bf1, acc[mi][1]);
    }
    __syncthreads();                       // tile consumed; safe to overwrite
  }

  int rhi = (lane >> 4) << 3;  // +8 rows for upper half-wave
  int nl  = lane & 15;
#pragma unroll
  for (int mi = 0; mi < 4; ++mi)
#pragma unroll
    for (int ni = 0; ni < 2; ++ni)
#pragma unroll
      for (int r = 0; r < 8; ++r) {
        int row = m0 + wm + mi * 16 + r + rhi;
        int col = n0 + wn + ni * 16 + nl;
        epi(row, col, acc[mi][ni][r]);
      }
}

// ---------------------------------------------------------------------------
// Epilogues
// ---------------------------------------------------------------------------
struct EpiQKV {   // cols [0,512)=q, [512,1024)=k, [1024,1536)=v
  bf16_t* Q;      // (b,h,n,d)
  bf16_t* KT;     // (b,h,d,n)
  bf16_t* VT;     // (b,h,d,n), pre-divided by N
  __device__ void operator()(int row, int col, float v) const {
    int b = row >> 12, n = row & (NN - 1);
    int part = col >> 9, cc = col & (CC - 1);
    int h = cc >> 5, d = cc & (DD - 1);
    size_t bh = (size_t)b * HH + h;
    if (part == 0) {
      float f = v > 0.f ? v + 1.f : expf(v);        // elu(v)+1
      Q[(bh * NN + n) * DD + d] = (bf16_t)f;
    } else if (part == 1) {
      float f = v > 0.f ? v + 1.f : expf(v);
      KT[(bh * DD + d) * NN + n] = (bf16_t)f;
    } else {
      VT[(bh * DD + d) * NN + n] = (bf16_t)(v * (1.f / (float)NN));
    }
  }
};

struct EpiWmRes {  // xnew = x + msg@Wm
  const float* x; float* xnew;
  __device__ void operator()(int row, int col, float v) const {
    size_t i = (size_t)row * CC + col;
    xnew[i] = x[i] + v;
  }
};

struct EpiW1Gelu { // f1 = gelu(h2@W1 + b1)  (exact gelu)
  const float* b1; bf16_t* f1;
  __device__ void operator()(int row, int col, float v) const {
    float t = v + b1[col];
    float g = 0.5f * t * (1.f + erff(t * 0.70710678118654752f));
    f1[(size_t)row * CC + col] = (bf16_t)g;
  }
};

struct EpiW2Out {  // out = xnew + f1@W2 + b2
  const float* b2; const float* xnew; float* out;
  __device__ void operator()(int row, int col, float v) const {
    size_t i = (size_t)row * CC + col;
    out[i] = xnew[i] + v + b2[col];
  }
};

// ---------------------------------------------------------------------------
// KV kernel: per (b,h) block computes KV[d][v] = sum_s K[s,d]*Vdiv[s,v]
// (M=32, N=32, K-dim=4096 via WMMA; waves split the K range), plus
// Ksum[d] = sum_s K[s,d]. Stores KV transposed (KVT[v][d], bf16).
// ---------------------------------------------------------------------------
__global__ __launch_bounds__(256) void kv_kernel(
    const bf16_t* __restrict__ KT, const bf16_t* __restrict__ VT,
    bf16_t* __restrict__ KVT, float* __restrict__ Ksum) {
  int bh = blockIdx.x;  // 0..127
  const bf16_t* Kt = KT + (size_t)bh * DD * NN;
  const bf16_t* Vt = VT + (size_t)bh * DD * NN;
  __shared__ float sKV[DD * DD];
  __shared__ float sKs[DD];
  int tid = threadIdx.x, wave = tid >> 5, lane = tid & 31;
  for (int i = tid; i < DD * DD; i += 256) sKV[i] = 0.f;
  if (tid < DD) sKs[tid] = 0.f;
  __syncthreads();

  f32x8 acc[2][2] = {};
#pragma unroll 4
  for (int it = 0; it < 16; ++it) {
    int s0 = wave * 512 + it * 32;
    bf16x16 a0 = load_frag(Kt, NN, 0, s0, lane);
    bf16x16 a1 = load_frag(Kt, NN, 16, s0, lane);
    bf16x16 b0 = load_frag(Vt, NN, 0, s0, lane);
    bf16x16 b1 = load_frag(Vt, NN, 16, s0, lane);
    acc[0][0] = wmma_bf16(a0, b0, acc[0][0]);
    acc[0][1] = wmma_bf16(a0, b1, acc[0][1]);
    acc[1][0] = wmma_bf16(a1, b0, acc[1][0]);
    acc[1][1] = wmma_bf16(a1, b1, acc[1][1]);
  }

  // Ksum partials: thread -> (d = tid>>3, s-chunk = tid&7 of 512)
  {
    int d = tid >> 3, chunk = tid & 7;
    const bf16_t* rowp = Kt + (size_t)d * NN + chunk * 512;
    float s = 0.f;
    for (int i = 0; i < 512; i += 8) {
      bf16x8 t = *(const bf16x8*)(rowp + i);
#pragma unroll
      for (int j = 0; j < 8; ++j) s += (float)t[j];
    }
    atomicAdd(&sKs[d], s);
  }

  int rhi = (lane >> 4) << 3;
  int nl = lane & 15;
#pragma unroll
  for (int mi = 0; mi < 2; ++mi)
#pragma unroll
    for (int ni = 0; ni < 2; ++ni)
#pragma unroll
      for (int r = 0; r < 8; ++r) {
        int dd = mi * 16 + r + rhi;
        int vv = ni * 16 + nl;
        atomicAdd(&sKV[dd * DD + vv], acc[mi][ni][r]);
      }
  __syncthreads();

  for (int i = tid; i < DD * DD; i += 256) {
    int d = i >> 5, vv = i & 31;
    KVT[(size_t)bh * (DD * DD) + vv * DD + d] = (bf16_t)sKV[i];  // transposed
  }
  if (tid < DD) Ksum[bh * DD + tid] = sKs[tid];
}

// ---------------------------------------------------------------------------
// msg kernel: per (b,h, row-tile of 512): num = Q@KV (M=512/blk, N=32, K=32),
// Z = 1/(Q . Ksum + eps), msg = num * Z * N -> bf16 (b,n,h,v) layout.
// ---------------------------------------------------------------------------
__global__ __launch_bounds__(256) void msg_kernel(
    const bf16_t* __restrict__ Q, const bf16_t* __restrict__ KVT,
    const float* __restrict__ Ksum, bf16_t* __restrict__ msg) {
  int bh = blockIdx.x; int b = bh >> 4, h = bh & 15;
  int r0 = blockIdx.y * 512;
  const bf16_t* Qh  = Q + (size_t)bh * NN * DD;
  const bf16_t* kvt = KVT + (size_t)bh * (DD * DD);
  __shared__ float sKs[DD];
  __shared__ float zbuf[512];
  int tid = threadIdx.x;
  if (tid < DD) sKs[tid] = Ksum[bh * DD + tid];
  __syncthreads();

  // Z per row: each thread handles 2 rows
#pragma unroll
  for (int i = 0; i < 2; ++i) {
    int lr = tid * 2 + i;
    const bf16_t* qr = Qh + (size_t)(r0 + lr) * DD;
    float dot = 0.f;
#pragma unroll
    for (int j = 0; j < DD; ++j) dot += (float)qr[j] * sKs[j];
    zbuf[lr] = 1.f / (dot + 1e-6f);
  }
  __syncthreads();

  int wave = tid >> 5, lane = tid & 31;
  int wm = wave * 64;
  bf16x16 bf0 = load_frag(kvt, DD, 0, 0, lane);
  bf16x16 bf1 = load_frag(kvt, DD, 16, 0, lane);
  f32x8 acc[4][2] = {};
#pragma unroll
  for (int mi = 0; mi < 4; ++mi) {
    bf16x16 af = load_frag(Qh, DD, r0 + wm + mi * 16, 0, lane);
    acc[mi][0] = wmma_bf16(af, bf0, acc[mi][0]);
    acc[mi][1] = wmma_bf16(af, bf1, acc[mi][1]);
  }

  int rhi = (lane >> 4) << 3;
  int nl = lane & 15;
#pragma unroll
  for (int mi = 0; mi < 4; ++mi)
#pragma unroll
    for (int ni = 0; ni < 2; ++ni)
#pragma unroll
      for (int r = 0; r < 8; ++r) {
        int lrow = wm + mi * 16 + r + rhi;
        int n = r0 + lrow;
        int vv = ni * 16 + nl;
        float val = acc[mi][ni][r] * zbuf[lrow] * (float)NN;
        msg[((size_t)(b * NN + n) * HH + h) * DD + vv] = (bf16_t)val;
      }
}

// ---------------------------------------------------------------------------
// Host launcher
// ---------------------------------------------------------------------------
extern "C" void kernel_launch(void* const* d_in, const int* in_sizes, int n_in,
                              void* d_out, int out_size, void* d_ws, size_t ws_size,
                              hipStream_t stream) {
  (void)in_sizes; (void)n_in; (void)out_size; (void)ws_size;
  const float* x     = (const float*)d_in[0];
  const float* Wq    = (const float*)d_in[1];
  const float* Wk    = (const float*)d_in[2];
  const float* Wv    = (const float*)d_in[3];
  const float* Wm    = (const float*)d_in[4];
  const float* W1    = (const float*)d_in[5];
  const float* b1    = (const float*)d_in[6];
  const float* W2    = (const float*)d_in[7];
  const float* b2    = (const float*)d_in[8];
  const float* g_att = (const float*)d_in[9];
  const float* b_att = (const float*)d_in[10];
  const float* g_ffn = (const float*)d_in[11];
  const float* b_ffn = (const float*)d_in[12];
  float* out = (float*)d_out;

  const size_t ROWS = (size_t)BB * NN;          // 32768
  char* wsb = (char*)d_ws;
  size_t off = 0;
  auto alloc = [&](size_t bytes) -> void* {
    off = (off + 255) & ~(size_t)255;
    void* p = wsb + off;
    off += bytes;
    return p;
  };
  bf16_t* WTqkv = (bf16_t*)alloc((size_t)3 * CC * CC * sizeof(bf16_t));
  bf16_t* WmT   = (bf16_t*)alloc((size_t)CC * CC * sizeof(bf16_t));
  bf16_t* W1T   = (bf16_t*)alloc((size_t)CC * CC * sizeof(bf16_t));
  bf16_t* W2T   = (bf16_t*)alloc((size_t)CC * CC * sizeof(bf16_t));
  bf16_t* hA    = (bf16_t*)alloc(ROWS * CC * sizeof(bf16_t));   // also reused as hF
  bf16_t* Qb    = (bf16_t*)alloc(ROWS * CC * sizeof(bf16_t));   // also reused as f1
  bf16_t* KTb   = (bf16_t*)alloc(ROWS * CC * sizeof(bf16_t));   // also reused as msg
  bf16_t* VTb   = (bf16_t*)alloc(ROWS * CC * sizeof(bf16_t));
  bf16_t* KVT   = (bf16_t*)alloc((size_t)BB * HH * DD * DD * sizeof(bf16_t));
  float*  Ksum  = (float*)alloc((size_t)BB * HH * DD * sizeof(float));
  float*  xnew  = (float*)alloc(ROWS * CC * sizeof(float));

  // 0. weight prep (f32 -> transposed bf16)
  prep_weights_kernel<<<(CC * CC) / 256, 256, 0, stream>>>(
      Wq, Wk, Wv, Wm, W1, W2, WTqkv, WmT, W1T, W2T);

  // 1. LN(att): x -> hA (bf16)
  ln_bf16_kernel<<<ROWS / 8, 256, 0, stream>>>(x, g_att, b_att, hA);

  // 2. fused QKV GEMM: hA @ [Wq|Wk|Wv], epilogue -> Q, KT, VT
  {
    EpiQKV epi{Qb, KTb, VTb};
    dim3 grid(ROWS / 128, (3 * CC) / 128);
    gemm_bf16_kernel<EpiQKV><<<grid, 256, 0, stream>>>(hA, WTqkv, CC, epi);
  }

  // 3. per-(b,h): KV = K^T @ Vdiv (WMMA, K=4096), Ksum
  kv_kernel<<<BB * HH, 256, 0, stream>>>(KTb, VTb, KVT, Ksum);

  // 4. msg = (Q @ KV) * Z * N  -> bf16 (b,n,h,v);  reuse KTb as msg buffer
  bf16_t* msg = KTb;
  msg_kernel<<<dim3(BB * HH, NN / 512), 256, 0, stream>>>(Qb, KVT, Ksum, msg);

  // 5. xnew = x + msg @ Wm
  {
    EpiWmRes epi{x, xnew};
    dim3 grid(ROWS / 128, CC / 128);
    gemm_bf16_kernel<EpiWmRes><<<grid, 256, 0, stream>>>(msg, WmT, CC, epi);
  }

  // 6. LN(ffn): xnew -> hF (reuse hA)
  bf16_t* hF = hA;
  ln_bf16_kernel<<<ROWS / 8, 256, 0, stream>>>(xnew, g_ffn, b_ffn, hF);

  // 7. f1 = gelu(hF @ W1 + b1) (reuse Qb)
  bf16_t* f1 = Qb;
  {
    EpiW1Gelu epi{b1, f1};
    dim3 grid(ROWS / 128, CC / 128);
    gemm_bf16_kernel<EpiW1Gelu><<<grid, 256, 0, stream>>>(hF, W1T, CC, epi);
  }

  // 8. out = xnew + f1 @ W2 + b2
  {
    EpiW2Out epi{b2, xnew, out};
    dim3 grid(ROWS / 128, CC / 128);
    gemm_bf16_kernel<EpiW2Out><<<grid, 256, 0, stream>>>(f1, W2T, CC, epi);
  }
}
